// S4Toy_10926396801047
// MI455X (gfx1250) — matrix operand order repaired and checked by reading, not compiled, not run on hardware
//
#include <hip/hip_runtime.h>

typedef __attribute__((ext_vector_type(16))) __bf16 v16bf;
typedef __attribute__((ext_vector_type(8)))  __bf16 v8bf;
typedef __attribute__((ext_vector_type(8)))  float  v8f;

#define Bsz 4
#define Lseq 2048
#define Hdim 1024
#define Nst 64
#define Mrows (Bsz * Lseq)   // 8192
#define Ncols (2 * Hdim)     // 2048

static __device__ inline v8f zero_v8f() {
    v8f z;
#pragma unroll
    for (int i = 0; i < 8; ++i) z[i] = 0.0f;
    return z;
}

// ---------------------------------------------------------------------------
// Kernel 0: W_out (H x 2H, f32, row-major) -> W^T (2H x H, bf16, row-major)
// so that GEMM B-staging reads contiguous K runs, same pattern as A.
// ---------------------------------------------------------------------------
__global__ __launch_bounds__(256) void s4_convert_wT(const float* __restrict__ W,
                                                     __bf16* __restrict__ WT) {
    int tid = blockIdx.x * 256 + threadIdx.x;   // 2M elements
    int n = tid >> 10;          // 0..2047
    int k = tid & 1023;         // 0..1023
    WT[tid] = (__bf16)W[k * Ncols + n];
}

// ---------------------------------------------------------------------------
// Kernel 1: S4D kernel generation.
// K[h,l] = 2*Re( sum_n Ct[h,n] * w_n^l ),  w_n = exp(dt*A_n),
// Ct = C*(w-1)/A. One block per h; thread t covers l = t*8 .. t*8+7 via a
// complex geometric recurrence seeded with exp/cos/sin at l0.
// ---------------------------------------------------------------------------
__global__ __launch_bounds__(256) void s4_genK(const float* __restrict__ A_re,
                                               const float* __restrict__ A_im,
                                               const float* __restrict__ log_dt,
                                               const float* __restrict__ C_re,
                                               const float* __restrict__ C_im,
                                               float* __restrict__ Kg) {
    const int h = blockIdx.x;
    const int t = threadIdx.x;
    const int l0 = t * 8;
    const float dt = __expf(log_dt[h]);
    float kacc[8];
#pragma unroll
    for (int i = 0; i < 8; ++i) kacc[i] = 0.0f;

    const float l0f = (float)l0;
    for (int n = 0; n < Nst; ++n) {
        const int idx = h * Nst + n;
        const float ar = A_re[idx], ai = A_im[idx];
        const float cr = C_re[idx], ci = C_im[idx];
        const float dar = ar * dt, dai = ai * dt;
        const float em = __expf(dar);
        const float er = em * __cosf(dai);     // w = exp(dt*A)
        const float ei = em * __sinf(dai);
        // Ct = C*(w-1)/A  (complex divide by A via conj)
        const float nr = cr * (er - 1.0f) - ci * ei;
        const float ni = cr * ei + ci * (er - 1.0f);
        const float inv = 1.0f / (ar * ar + ai * ai);
        const float ctr = (nr * ar + ni * ai) * inv;
        const float cti = (ni * ar - nr * ai) * inv;
        // p = Ct * w^l0, argument-reduced
        const float el = __expf(dar * l0f);
        const float th = fmodf(dai * l0f, 6.28318530718f);
        const float wr = el * __cosf(th), wi = el * __sinf(th);
        float pr = ctr * wr - cti * wi;
        float pi = ctr * wi + cti * wr;
#pragma unroll
        for (int i = 0; i < 8; ++i) {
            kacc[i] += 2.0f * pr;
            const float npr = pr * er - pi * ei;
            pi = pr * ei + pi * er;
            pr = npr;
        }
    }
    float4* out = (float4*)(Kg + h * Lseq + l0);
    out[0] = make_float4(kacc[0], kacc[1], kacc[2], kacc[3]);
    out[1] = make_float4(kacc[4], kacc[5], kacc[6], kacc[7]);
}

// ---------------------------------------------------------------------------
// Kernel 2: causal convolution + skip + GELU, output bf16 activations.
// Block = one (h, b); K row and x column live in LDS (16 KB). Each thread
// handles a low window [t*4, t*4+3] and a mirrored high window so per-thread
// work (proportional to l) is balanced across the block.
// ---------------------------------------------------------------------------
static __device__ inline float conv_at(int l, const float* Ks, const float* xs) {
    float acc = 0.0f;
    int j = 0;
    for (; j + 3 <= l; j += 4) {                 // Ks 16B-aligned chunks
        const float4 kq = *(const float4*)(Ks + j);
        acc += kq.x * xs[l - j] + kq.y * xs[l - j - 1] +
               kq.z * xs[l - j - 2] + kq.w * xs[l - j - 3];
    }
    for (; j <= l; ++j) acc += Ks[j] * xs[l - j];
    return acc;
}

static __device__ inline float gelu_tanh(float y) {
    const float c = 0.7978845608028654f;
    return 0.5f * y * (1.0f + tanhf(c * (y + 0.044715f * y * y * y)));
}

__global__ __launch_bounds__(256) void s4_conv(const float* __restrict__ x,
                                               const float* __restrict__ Kg,
                                               const float* __restrict__ Dvec,
                                               __bf16* __restrict__ G) {
    __shared__ __align__(16) float Ks[Lseq];
    __shared__ __align__(16) float xs[Lseq];
    const int h = blockIdx.x;
    const int b = blockIdx.y;
    const int t = threadIdx.x;
#pragma unroll
    for (int i = 0; i < 8; ++i) {
        const int l = t + 256 * i;
        Ks[l] = Kg[h * Lseq + l];
        xs[l] = x[(b * Lseq + l) * Hdim + h];
    }
    __syncthreads();
    const float Dh = Dvec[h];
#pragma unroll
    for (int i = 0; i < 4; ++i) {
        {   // low window: l in [0, 1023]
            const int l = t * 4 + i;
            const float y = conv_at(l, Ks, xs) + xs[l] * Dh;
            G[(b * Lseq + l) * Hdim + h] = (__bf16)gelu_tanh(y);
        }
        {   // high window: l in [1024, 2047], mirrored for load balance
            const int l = (Lseq - 4) - t * 4 + i;
            const float y = conv_at(l, Ks, xs) + xs[l] * Dh;
            G[(b * Lseq + l) * Hdim + h] = (__bf16)gelu_tanh(y);
        }
    }
}

// ---------------------------------------------------------------------------
// Kernel 3: GEMM (8192 x 1024) @ (1024 x 2048) with v_wmma_f32_16x16x32_bf16,
// double-buffered LDS staging and fully in-register bias+GLU epilogue.
//
// Block tile: 128 rows x (64 u-cols + 64 v-cols). 8 waves = 4(M) x 2(N),
// wave tile 32 rows x (32 u + 32 v): 2 A-frags x 4 B-frags = 8 WMMAs/k-step.
// u-frag ni and v-frag ni+2 share the identical (M,N) lane mapping, so GLU
// is elementwise in registers -> direct coalesced global stores.
// A-frag per ISA: lane<16 -> K{0..7,16..23}; lane>=16 -> K{8..15,24..31}.
// B-frag per ISA: lane<16 -> K{0..7,16..23}; lane>=16 -> K{8..15,24..31}.
// ---------------------------------------------------------------------------
static __device__ inline v16bf load_afrag(const __bf16* row, int half) {
    const v8bf lo = *(const v8bf*)(row + half * 8);
    const v8bf hi = *(const v8bf*)(row + half * 8 + 16);
    v16bf f;
#pragma unroll
    for (int i = 0; i < 8; ++i) { f[i] = lo[i]; f[i + 8] = hi[i]; }
    return f;
}

static __device__ inline v16bf load_bfrag(const __bf16* row, int half) {
    const v8bf lo = *(const v8bf*)(row + half * 8);
    const v8bf hi = *(const v8bf*)(row + half * 8 + 16);
    v16bf f;
#pragma unroll
    for (int i = 0; i < 8; ++i) { f[i] = lo[i]; f[i + 8] = hi[i]; }
    return f;
}

#define A_STRIDE 40                 // 32 k + 8 pad (16B-aligned rows)
#define TBUF (128 * A_STRIDE)       // one 128-row tile buffer (5120 elems)

__global__ __launch_bounds__(256) void s4_gemm_glu(const __bf16* __restrict__ G,
                                                   const __bf16* __restrict__ WT,
                                                   const float* __restrict__ bout,
                                                   float* __restrict__ Y2) {
    // A0 | A1 | B0 | B1  (double-buffered) = 40960 B LDS
    __shared__ __align__(16) __bf16 smem[4 * TBUF];

    const int t    = threadIdx.x;
    const int lane = t & 31;
    const int wave = t >> 5;
    const int half = lane >> 4;
    const int lm   = lane & 15;
    const int m0 = blockIdx.y * 128;
    const int n0 = blockIdx.x * 64;
    const int wm = (wave & 3) * 32;     // 4 M-tiles of 32
    const int wn = (wave >> 2) * 32;    // 2 N-tiles of 32 (u part)

    v8f acc[2][4];
#pragma unroll
    for (int mi = 0; mi < 2; ++mi)
#pragma unroll
        for (int ni = 0; ni < 4; ++ni) acc[mi][ni] = zero_v8f();

    // staging: thread covers (row = t>>1, 16 k's at (t&1)*16) for both tiles
    const int sr = t >> 1;            // 0..127
    const int sk = (t & 1) * 16;      // 0,16
    const int bng = (sr < 64) ? (n0 + sr) : (Hdim + n0 + (sr - 64));
    const __bf16* gA = G + (m0 + sr) * Hdim + sk;
    const __bf16* gB = WT + bng * Hdim + sk;
    const int soff = sr * A_STRIDE + sk;

    // prologue: stage k-tile 0 into buffer 0
    {
        const v8bf a0 = *(const v8bf*)(gA);
        const v8bf a1 = *(const v8bf*)(gA + 8);
        const v8bf b0 = *(const v8bf*)(gB);
        const v8bf b1 = *(const v8bf*)(gB + 8);
        *(v8bf*)(smem + soff) = a0;
        *(v8bf*)(smem + soff + 8) = a1;
        *(v8bf*)(smem + 2 * TBUF + soff) = b0;
        *(v8bf*)(smem + 2 * TBUF + soff + 8) = b1;
    }
    __syncthreads();

    int cur = 0;
    for (int it = 0; it < Hdim / 32; ++it) {
        const int kn = (it + 1) * 32;
        const bool more = (kn < Hdim);
        v8bf a0, a1, b0, b1;
        if (more) {     // issue next-tile global loads before the WMMA burst
            a0 = *(const v8bf*)(gA + kn);
            a1 = *(const v8bf*)(gA + kn + 8);
            b0 = *(const v8bf*)(gB + kn);
            b1 = *(const v8bf*)(gB + kn + 8);
            if (kn + 32 < Hdim) {
                __builtin_prefetch(gA + kn + 32, 0, 1);
                __builtin_prefetch(gB + kn + 32, 0, 1);
            }
        }

        const __bf16* Ac = smem + cur * TBUF;
        const __bf16* Bc = smem + (2 + cur) * TBUF;
        v16bf afr[2], bfr[4];
#pragma unroll
        for (int mi = 0; mi < 2; ++mi)
            afr[mi] = load_afrag(Ac + (wm + 16 * mi + lm) * A_STRIDE, half);
#pragma unroll
        for (int ni = 0; ni < 4; ++ni) {
            const int brow = (ni < 2) ? (wn + 16 * ni)
                                      : (64 + wn + 16 * (ni - 2));
            bfr[ni] = load_bfrag(Bc + (brow + lm) * A_STRIDE, half);
        }
#pragma unroll
        for (int mi = 0; mi < 2; ++mi)
#pragma unroll
            for (int ni = 0; ni < 4; ++ni)
                acc[mi][ni] = __builtin_amdgcn_wmma_f32_16x16x32_bf16(
                    false, afr[mi], false, bfr[ni], (short)0, acc[mi][ni],
                    false, false);

        if (more) {     // fill the alternate buffer, single barrier per step
            __bf16* An = smem + (cur ^ 1) * TBUF + soff;
            __bf16* Bn = smem + (2 + (cur ^ 1)) * TBUF + soff;
            *(v8bf*)(An) = a0;
            *(v8bf*)(An + 8) = a1;
            *(v8bf*)(Bn) = b0;
            *(v8bf*)(Bn + 8) = b1;
            __syncthreads();
            cur ^= 1;
        }
    }

    // in-register bias + GLU epilogue: u = acc[mi][ni], v = acc[mi][ni+2]
    // element r -> row m0+wm+16*mi+8*half+r, col n0+wn+16*ni+lm
#pragma unroll
    for (int mi = 0; mi < 2; ++mi) {
#pragma unroll
        for (int ni = 0; ni < 2; ++ni) {
            const int col = n0 + wn + 16 * ni + lm;
            const float bu = bout[col];
            const float bv = bout[Hdim + col];
            const int rowbase = m0 + wm + 16 * mi + 8 * half;
#pragma unroll
            for (int r = 0; r < 8; ++r) {
                const float u = acc[mi][ni][r] + bu;
                const float v = acc[mi][ni + 2][r] + bv;
                const float s = 1.0f / (1.0f + __expf(-v));
                Y2[(rowbase + r) * Hdim + col] = u * s;
            }
        }
    }
}

// ---------------------------------------------------------------------------
// Kernel 4: LayerNorm over H=1024 per row, scale/shift, write final f32 out.
// ---------------------------------------------------------------------------
__global__ __launch_bounds__(256) void s4_ln(const float* __restrict__ Y2,
                                             const float* __restrict__ gamma,
                                             const float* __restrict__ beta,
                                             float* __restrict__ out) {
    __shared__ float s1[256];
    __shared__ float s2[256];
    const int m = blockIdx.x;
    const int t = threadIdx.x;
    const float4 v = *(const float4*)(Y2 + m * Hdim + t * 4);
    s1[t] = v.x + v.y + v.z + v.w;
    s2[t] = v.x * v.x + v.y * v.y + v.z * v.z + v.w * v.w;
    __syncthreads();
    for (int s = 128; s > 0; s >>= 1) {
        if (t < s) { s1[t] += s1[t + s]; s2[t] += s2[t + s]; }
        __syncthreads();
    }
    const float mu = s1[0] * (1.0f / Hdim);
    const float var = s2[0] * (1.0f / Hdim) - mu * mu;
    const float rinv = rsqrtf(var + 1e-5f);
    const int c = t * 4;
    float4 o;
    o.x = (v.x - mu) * rinv * gamma[c + 0] + beta[c + 0];
    o.y = (v.y - mu) * rinv * gamma[c + 1] + beta[c + 1];
    o.z = (v.z - mu) * rinv * gamma[c + 2] + beta[c + 2];
    o.w = (v.w - mu) * rinv * gamma[c + 3] + beta[c + 3];
    *(float4*)(out + m * Hdim + c) = o;
}

// ---------------------------------------------------------------------------
extern "C" void kernel_launch(void* const* d_in, const int* in_sizes, int n_in,
                              void* d_out, int out_size, void* d_ws, size_t ws_size,
                              hipStream_t stream) {
    const float* x      = (const float*)d_in[0];
    const float* A_re   = (const float*)d_in[1];
    const float* A_im   = (const float*)d_in[2];
    const float* log_dt = (const float*)d_in[3];
    const float* C_re   = (const float*)d_in[4];
    const float* C_im   = (const float*)d_in[5];
    const float* Dvec   = (const float*)d_in[6];
    const float* W_out  = (const float*)d_in[7];
    const float* b_out  = (const float*)d_in[8];
    const float* gamma  = (const float*)d_in[9];
    const float* beta   = (const float*)d_in[10];

    char* ws = (char*)d_ws;
    float*  Kg = (float*)ws;                                  // 8 MB
    __bf16* G  = (__bf16*)(ws + (size_t)(8u << 20));          // 16 MB
    __bf16* WT = (__bf16*)(ws + (size_t)(24u << 20));         // 4 MB
    float*  Y2 = (float*)(ws + (size_t)(28u << 20));          // 32 MB

    s4_convert_wT<<<(Ncols * Hdim) / 256, 256, 0, stream>>>(W_out, WT);
    s4_genK<<<Hdim, 256, 0, stream>>>(A_re, A_im, log_dt, C_re, C_im, Kg);
    s4_conv<<<dim3(Hdim, Bsz), 256, 0, stream>>>(x, Kg, Dvec, G);
    s4_gemm_glu<<<dim3(Hdim / 64, Mrows / 128), 256, 0, stream>>>(G, WT, b_out, Y2);
    s4_ln<<<Mrows, 256, 0, stream>>>(Y2, gamma, beta, (float*)d_out);
}
